// R2Ntab_16295105921423
// MI455X (gfx1250) — matrix operand order repaired and checked by reading, not compiled, not run on hardware
//
#include <hip/hip_runtime.h>

typedef __attribute__((ext_vector_type(16))) _Float16 v16h;
typedef __attribute__((ext_vector_type(8)))  _Float16 v8h;
typedef __attribute__((ext_vector_type(2)))  __fp16   v2fp16;
typedef __attribute__((ext_vector_type(8)))  float    v8f;
typedef __attribute__((ext_vector_type(4)))  float    fvec4;
typedef __attribute__((ext_vector_type(4)))  unsigned int uvec4;

// async-to-LDS builtin parameter types (per hipcc diagnostic:
// "int __vector_size(16) __device__ *")
typedef int vsi4 __attribute__((vector_size(16)));
typedef vsi4 __attribute__((address_space(1)))* async_gptr;
typedef vsi4 __attribute__((address_space(3)))* async_lptr;

#define B_SZ 131072
#define F_SZ 256
#define R_SZ 128
#define LDSW 264   // halves per LDS row (256 + 8 pad -> conflict-free b128 reads)

#if __has_builtin(__builtin_amdgcn_global_load_async_to_lds_b128)
#define USE_ASYNC_LDS 1
#else
#define USE_ASYNC_LDS 0
#endif

// ---------------------------------------------------------------------------
// Prep: rw[r,f] = (w_cancel[f] < 0) ? 0 : w_and[r,f]   (stored as f16)
//       crule[r] = b_and[r] - sum_f relu(rw[r,f])      (f16-consistent)
// One block per rule, 256 threads (= F).
// ---------------------------------------------------------------------------
__global__ __launch_bounds__(256)
void r2ntab_prep(const float* __restrict__ w_cancel,
                 const float* __restrict__ w_and,
                 const float* __restrict__ b_and,
                 _Float16* __restrict__ w16,
                 float* __restrict__ crule) {
  const int r = blockIdx.x;
  const int f = threadIdx.x;
  const float wc = w_cancel[f];
  const float wa = w_and[r * F_SZ + f];
  const float rw = (wc < 0.0f) ? 0.0f : wa;
  const _Float16 h16 = (_Float16)rw;
  w16[r * F_SZ + f] = h16;
  const float rwf = (float)h16;
  float pos = rwf > 0.0f ? rwf : 0.0f;
  #pragma unroll
  for (int m = 16; m; m >>= 1) pos += __shfl_xor(pos, m, 32);
  __shared__ float red[8];
  if ((threadIdx.x & 31) == 0) red[threadIdx.x >> 5] = pos;
  __syncthreads();
  if (threadIdx.x == 0) {
    float s = 0.0f;
    #pragma unroll
    for (int i = 0; i < 8; ++i) s += red[i];
    crule[r] = b_and[r] - s;
  }
}

// ---------------------------------------------------------------------------
// Main: per block, stage 128 rows of x (f32 -> f16, packed cvt, NT loads) and
// the f16 weights (async global->LDS) into LDS, then 8 waves each run a
// 16x128 WMMA strip, threshold, and reduce.
// ---------------------------------------------------------------------------
__global__ __launch_bounds__(256)
void r2ntab_main(const float* __restrict__ x,
                 _Float16* __restrict__ w16,
                 const float* __restrict__ crule,
                 const float* __restrict__ w_or,
                 const float* __restrict__ b_or,
                 float* __restrict__ out) {
  extern __shared__ char smem[];
  _Float16* xs  = (_Float16*)smem;          // [128][LDSW]
  _Float16* wsh = xs + 128 * LDSW;          // [128][LDSW]

  const int tid = threadIdx.x;
  const int rowBlock = blockIdx.x * 128;

  // ---- stage weights: 128 x 256 halves, async global -> LDS ----
  {
    vsi4* wg = (vsi4*)w16;
    #pragma unroll
    for (int i = 0; i < 16; ++i) {
      const int idx = tid + i * 256;        // 0..4095 16B chunks (8 halves)
      const int row = idx >> 5;             // 32 chunks per row
      const int c8  = idx & 31;
      _Float16* dst = wsh + row * LDSW + c8 * 8;
#if USE_ASYNC_LDS
      __builtin_amdgcn_global_load_async_to_lds_b128(
          (async_gptr)(uintptr_t)(wg + idx),
          (async_lptr)(uintptr_t)dst, 0, 0);
#else
      *(vsi4*)dst = wg[idx];
#endif
    }
  }

  // ---- stage x tile: 128 x 256 f32 -> f16, 8 halves (one b128) per iter ----
  {
    const fvec4* xg = (const fvec4*)(x + (size_t)rowBlock * F_SZ);
    #pragma unroll
    for (int i = 0; i < 16; ++i) {
      const int idx = tid + i * 256;        // 0..4095 chunks of 8 floats
      const int row = idx >> 5;             // 32 chunks per row
      const int c8  = idx & 31;
      const fvec4 v0 = __builtin_nontemporal_load(xg + 2 * idx);
      const fvec4 v1 = __builtin_nontemporal_load(xg + 2 * idx + 1);
      union { struct { v2fp16 a, b, c, d; } s; uvec4 u; } pk;
#if __has_builtin(__builtin_amdgcn_cvt_pkrtz)
      pk.s.a = __builtin_amdgcn_cvt_pkrtz(v0.x, v0.y);   // exact: x in {0,1}
      pk.s.b = __builtin_amdgcn_cvt_pkrtz(v0.z, v0.w);
      pk.s.c = __builtin_amdgcn_cvt_pkrtz(v1.x, v1.y);
      pk.s.d = __builtin_amdgcn_cvt_pkrtz(v1.z, v1.w);
#else
      pk.s.a = (v2fp16){(__fp16)v0.x, (__fp16)v0.y};
      pk.s.b = (v2fp16){(__fp16)v0.z, (__fp16)v0.w};
      pk.s.c = (v2fp16){(__fp16)v1.x, (__fp16)v1.y};
      pk.s.d = (v2fp16){(__fp16)v1.z, (__fp16)v1.w};
#endif
      *(uvec4*)(xs + row * LDSW + c8 * 8) = pk.u;
    }
  }

#if USE_ASYNC_LDS
#if __has_builtin(__builtin_amdgcn_s_wait_asynccnt)
  __builtin_amdgcn_s_wait_asynccnt(0);
#else
  asm volatile("s_wait_asynccnt 0x0" ::: "memory");
#endif
#endif
  __syncthreads();

  const int wave = tid >> 5;
  const int lane = tid & 31;
  const int lm = lane & 15;
  const int lh = lane >> 4;
  const int rowLoc = wave * 16;             // wave's 16-row strip

  float cnt[8];
  #pragma unroll
  for (int v = 0; v < 8; ++v) cnt[v] = 0.0f;

  const float bor = b_or[0];

  // A-frag base: row (rowLoc+lm), K elements {lh*8+0..7, lh*8+16..23} per step
  const _Float16* abase = xs + (rowLoc + lm) * LDSW + lh * 8;

  #pragma unroll
  for (int nt = 0; nt < 8; ++nt) {          // 8 rule tiles of 16
    v8f acc = {};
    // B-frag base: column rule = nt*16+lm, K elements {lh*8+0..7, lh*8+16..23} per step
    const _Float16* bbase = wsh + (nt * 16 + lm) * LDSW + lh * 8;
    #pragma unroll
    for (int ks = 0; ks < 8; ++ks) {        // K = 256 in steps of 32
      union { v16h v; v8h h[2]; } A, Bm;
      A.h[0]  = *(const v8h*)(abase + ks * 32);
      A.h[1]  = *(const v8h*)(abase + ks * 32 + 16);
      Bm.h[0] = *(const v8h*)(bbase + ks * 32);
      Bm.h[1] = *(const v8h*)(bbase + ks * 32 + 16);
      acc = __builtin_amdgcn_wmma_f32_16x16x32_f16(
          /*neg_a=*/false, A.v, /*neg_b=*/false, Bm.v,
          /*c_mod=*/(short)0, acc, /*reuse_a=*/false, /*reuse_b=*/false);
    }
    const int rule = nt * 16 + lm;
    const float cv = crule[rule];
    const float wb = (w_or[rule] > 0.0f) ? 1.0f : 0.0f;
    #pragma unroll
    for (int v = 0; v < 8; ++v) {
      const float h = acc[v] + cv;          // = xc@rw.T + b_and - relu_sum
      cnt[v] += (h > 0.999999f) ? wb : 0.0f;
    }
  }

  // reduce fired-counts across the 16-lane rule groups (xor 1,2,4,8 stays
  // within each half of the wave32)
  #pragma unroll
  for (int v = 0; v < 8; ++v) {
    float s = cnt[v];
    s += __shfl_xor(s, 1, 32);
    s += __shfl_xor(s, 2, 32);
    s += __shfl_xor(s, 4, 32);
    s += __shfl_xor(s, 8, 32);
    cnt[v] = s;
  }
  if (lm == 0) {
    #pragma unroll
    for (int v = 0; v < 8; ++v)
      out[rowBlock + rowLoc + v + 8 * lh] = cnt[v] + bor;  // O == 1
  }
}

// ---------------------------------------------------------------------------
extern "C" void kernel_launch(void* const* d_in, const int* in_sizes, int n_in,
                              void* d_out, int out_size, void* d_ws, size_t ws_size,
                              hipStream_t stream) {
  const float* x        = (const float*)d_in[0];  // [B,F]
  const float* w_cancel = (const float*)d_in[1];  // [F]
  const float* w_and    = (const float*)d_in[2];  // [R,F]
  const float* b_and    = (const float*)d_in[3];  // [R]
  const float* w_or     = (const float*)d_in[4];  // [O,R]
  const float* b_or     = (const float*)d_in[5];  // [O]

  _Float16* w16 = (_Float16*)d_ws;                            // 64 KB
  float* crule  = (float*)((char*)d_ws + R_SZ * F_SZ * sizeof(_Float16));

  r2ntab_prep<<<R_SZ, F_SZ, 0, stream>>>(w_cancel, w_and, b_and, w16, crule);

  const size_t shmem = (size_t)2 * 128 * LDSW * sizeof(_Float16);  // 132 KB
  r2ntab_main<<<B_SZ / 128, 256, shmem, stream>>>(
      x, w16, crule, w_or, b_or, (float*)d_out);
}
